// SelfAttention_2826088481408
// MI455X (gfx1250) — matrix operand order repaired and checked by reading, not compiled, not run on hardware
//
#include <hip/hip_runtime.h>
#include <math.h>

typedef _Float16 h16;
typedef __attribute__((ext_vector_type(16))) _Float16 v16h;
typedef __attribute__((ext_vector_type(8)))  _Float16 v8h;
typedef __attribute__((ext_vector_type(8)))  float    v8f;

#define DEV static __device__ __forceinline__

// ---------------------------------------------------------------------------
// Fragment loaders (wave32, CDNA5 WMMA layouts, ISA 7.12.2)
// A (16-bit, 16x32, MxK): lane L holds row m=L%16.
//   halves e=0..7  -> k = (L<16?0:8)  + e
//   halves e=8..15 -> k = 16 + (L<16?0:8) + (e-8)
// B (16-bit, 32x16, KxN): lane L holds column n=L%16.
//   halves e=0..15 -> same k order as A (e<8: koff+e, e>=8: 16+koff+e-8)
// C/D (f32 16x16): vgpr r: m = r + (L<16?0:8), n = L%16
// ---------------------------------------------------------------------------

DEV v16h frag_a_f16(const h16* __restrict__ base, int row0, int ld) {
    int lane = threadIdx.x & 31;
    const h16* p = base + (size_t)(row0 + (lane & 15)) * ld + ((lane & 16) ? 8 : 0);
    v8h lo = *(const v8h*)p;        // k = koff .. koff+7
    v8h hi = *(const v8h*)(p + 16); // k = 16+koff .. 16+koff+7
    v16h r;
#pragma unroll
    for (int i = 0; i < 8; ++i) { r[i] = lo[i]; r[i + 8] = hi[i]; }
    return r;
}

DEV v16h frag_b_f16(const h16* __restrict__ base, int col0, int ld) {
    // source stored "B-transposed row-major": row index = n, k contiguous
    int lane = threadIdx.x & 31;
    const h16* p = base + (size_t)(col0 + (lane & 15)) * ld + ((lane & 16) ? 8 : 0);
    v8h lo = *(const v8h*)p;        // k = koff .. koff+7
    v8h hi = *(const v8h*)(p + 16); // k = 16+koff .. 16+koff+7
    v16h r;
#pragma unroll
    for (int i = 0; i < 8; ++i) { r[i] = lo[i]; r[i + 8] = hi[i]; }
    return r;
}

DEV v16h frag_a_f32(const float* __restrict__ base, int row0, int ld, int k0) {
    int lane = threadIdx.x & 31;
    const float* p = base + (size_t)(row0 + (lane & 15)) * ld + k0 + ((lane & 16) ? 8 : 0);
    float4 a = *(const float4*)(p);
    float4 b = *(const float4*)(p + 4);
    float4 c = *(const float4*)(p + 16);
    float4 d = *(const float4*)(p + 20);
    v16h r;
    r[0]  = (h16)a.x; r[1]  = (h16)a.y; r[2]  = (h16)a.z; r[3]  = (h16)a.w;
    r[4]  = (h16)b.x; r[5]  = (h16)b.y; r[6]  = (h16)b.z; r[7]  = (h16)b.w;
    r[8]  = (h16)c.x; r[9]  = (h16)c.y; r[10] = (h16)c.z; r[11] = (h16)c.w;
    r[12] = (h16)d.x; r[13] = (h16)d.y; r[14] = (h16)d.z; r[15] = (h16)d.w;
    return r;
}

DEV v8f wmma_f16(v16h a, v16h b, v8f c) {
    return __builtin_amdgcn_wmma_f32_16x16x32_f16(false, a, false, b, (short)0, c,
                                                  false, false);
}

// ---------------------------------------------------------------------------
// Problem constants
// ---------------------------------------------------------------------------
#define BATCH   16
#define HW      4096
#define CIN     256
#define DK      32    // C/8
#define DV      128   // C/2
#define NPIX    (BATCH * HW)   // 65536

// ---------------------------------------------------------------------------
// 0) weight transpose + f16 convert:  dst[n*K + k] = (h16) src[k*N + n]
// ---------------------------------------------------------------------------
__global__ void transpose_cvt_kernel(const float* __restrict__ src,
                                     h16* __restrict__ dst, int K, int N) {
    int idx = blockIdx.x * blockDim.x + threadIdx.x;
    if (idx >= K * N) return;
    int k = idx / N, n = idx - k * N;
    dst[(size_t)n * K + k] = (h16)src[idx];
}

// ---------------------------------------------------------------------------
// 1) fused projections: one wave computes a 16-pixel tile x all 192 channels.
//    wt: [192][256] f16 rows 0-31 Wf^T, 32-63 Wg^T, 64-191 Wh^T
//    fh,gh: [65536][32] f16 row-major;  ht: [16][128][4096] f16 (transposed)
// ---------------------------------------------------------------------------
__global__ __launch_bounds__(256) void proj_kernel(
    const float* __restrict__ x, const h16* __restrict__ wt,
    const float* __restrict__ bf, const float* __restrict__ bg,
    const float* __restrict__ bh, h16* __restrict__ fh, h16* __restrict__ gh,
    h16* __restrict__ ht) {
    int wave = blockIdx.x * (blockDim.x >> 5) + (threadIdx.x >> 5);
    int row0 = wave * 16;                 // pixel tile base
    int lane  = threadIdx.x & 31;
    int rbase = (lane & 16) ? 8 : 0;
    int n     = lane & 15;

    v16h a[8];
#pragma unroll
    for (int kc = 0; kc < 8; ++kc) a[kc] = frag_a_f32(x, row0, CIN, kc * 32);

    for (int nt = 0; nt < 12; ++nt) {
        v8f acc = {};
#pragma unroll
        for (int kc = 0; kc < 8; ++kc) {
            v16h b = frag_b_f16(wt + kc * 32, nt * 16, CIN);
            acc = wmma_f16(a[kc], b, acc);
        }
#pragma unroll
        for (int r = 0; r < 8; ++r) {
            int prow = row0 + rbase + r;
            if (nt < 2) {
                int ch = nt * 16 + n;
                fh[(size_t)prow * DK + ch] = (h16)(acc[r] + bf[ch]);
            } else if (nt < 4) {
                int ch = (nt - 2) * 16 + n;
                gh[(size_t)prow * DK + ch] = (h16)(acc[r] + bg[ch]);
            } else {
                int ch = (nt - 4) * 16 + n;
                int b_ = prow >> 12, pos = prow & (HW - 1);
                ht[((size_t)b_ * DV + ch) * HW + pos] = (h16)(acc[r] + bh[ch]);
            }
        }
    }
}

// ---------------------------------------------------------------------------
// 2) flash attention, block-cooperative K/V staging.
//    Block = 8 waves = 128 queries of one batch. K/V chunk (32 keys) is staged
//    once into LDS (double buffered) and shared by all 8 waves: 8x less L2
//    read traffic, and staging overlaps the 10 WMMAs of the previous chunk.
//    o_ws: [16][4096][128] f32 (unprojected attention output)
// ---------------------------------------------------------------------------
__global__ __launch_bounds__(256) void attn_kernel(
    const h16* __restrict__ fh, const h16* __restrict__ gh,
    const h16* __restrict__ ht, float* __restrict__ o_ws) {
    __shared__ h16 sK[2][32 * 32];    // [buf][key][d]   4 KB
    __shared__ h16 sV[2][128 * 32];   // [buf][ch][key] 16 KB
    __shared__ h16 pbuf[8 * 16 * 32]; // per-wave P staging 8 KB

    int tid    = threadIdx.x;
    int waveId = tid >> 5;
    int lane   = tid & 31;
    int gt     = blockIdx.x * 8 + waveId; // global query tile (0..4095)
    int b      = blockIdx.x >> 5;         // 32 blocks per batch (8 tiles each)
    int q0     = (gt & 255) * 16;
    const h16* fb = fh + (size_t)b * HW * DK;
    const h16* gb = gh + (size_t)b * HW * DK;
    const h16* hb = ht + (size_t)b * DV * HW;
    h16* sp = pbuf + waveId * (16 * 32);

    int rbase = (lane & 16) ? 8 : 0;
    int n     = lane & 15;

    v16h q = frag_a_f16(gb, q0, DK);
    v8f  oacc[8];
    float mrow[8], lrow[8];
#pragma unroll
    for (int t = 0; t < 8; ++t) { v8f z = {}; oacc[t] = z; }
#pragma unroll
    for (int r = 0; r < 8; ++r) { mrow[r] = -INFINITY; lrow[r] = 0.0f; }

    // cooperative staging of one 32-key chunk: K 2KB (64 thr), V 8KB (256 thr)
    auto stage = [&](int kt, int buf) {
        if (tid < 64) {
            int key = tid >> 1, hh = (tid & 1) * 16;
            h16* dk = &sK[buf][key * 32 + hh];
            const h16* sk = fb + (size_t)(kt + key) * DK + hh;
            *(v8h*)dk       = *(const v8h*)sk;
            *(v8h*)(dk + 8) = *(const v8h*)(sk + 8);
        }
        int ch = tid >> 1, hh = (tid & 1) * 16;
        h16* dv = &sV[buf][ch * 32 + hh];
        const h16* sv = hb + (size_t)ch * HW + kt + hh;
        *(v8h*)dv       = *(const v8h*)sv;
        *(v8h*)(dv + 8) = *(const v8h*)(sv + 8);
    };

    stage(0, 0);
    __syncthreads();

    int buf = 0;
    for (int kt = 0; kt < HW; kt += 32, buf ^= 1) {
        if (kt + 32 < HW) stage(kt + 32, buf ^ 1);
        if (kt + 64 < HW) {
            // pull chunk kt+64 into cache while computing: global_prefetch_b8
            if (tid < 128)
                __builtin_prefetch(hb + (size_t)tid * HW + kt + 64, 0, 1);
            else if (tid < 144)
                __builtin_prefetch(fb + (size_t)(kt + 64) * DK + (tid - 128) * 64,
                                   0, 1);
        }

        const h16* sKc = sK[buf];
        const h16* sVc = sV[buf];
        v16h k0 = frag_b_f16(sKc, 0, 32);
        v16h k1 = frag_b_f16(sKc, 16, 32);
        v8f  z  = {};
        v8f  s0 = wmma_f16(q, k0, z);
        v8f  s1 = wmma_f16(q, k1, z);
#pragma unroll
        for (int r = 0; r < 8; ++r) {
            // row max over 32 keys (butterfly within the 16-lane D-layout group)
            float mx = fmaxf(s0[r], s1[r]);
            mx = fmaxf(mx, __shfl_xor(mx, 1, 32));
            mx = fmaxf(mx, __shfl_xor(mx, 2, 32));
            mx = fmaxf(mx, __shfl_xor(mx, 4, 32));
            mx = fmaxf(mx, __shfl_xor(mx, 8, 32));
            float mn    = fmaxf(mrow[r], mx);
            float scale = __expf(mrow[r] - mn);
            float p0 = __expf(s0[r] - mn);
            float p1 = __expf(s1[r] - mn);
            float rs = p0 + p1;
            rs += __shfl_xor(rs, 1, 32);
            rs += __shfl_xor(rs, 2, 32);
            rs += __shfl_xor(rs, 4, 32);
            rs += __shfl_xor(rs, 8, 32);
            lrow[r] = lrow[r] * scale + rs;
            mrow[r] = mn;
#pragma unroll
            for (int t = 0; t < 8; ++t) oacc[t][r] *= scale;
            int mi = rbase + r;                      // D-layout row
            sp[mi * 32 + n]      = (h16)p0;
            sp[mi * 32 + n + 16] = (h16)p1;
        }
        // reshape P (D-layout -> A-layout) via per-wave LDS round-trip
        v16h pf = frag_a_f16(sp, 0, 32);
#pragma unroll
        for (int t = 0; t < 8; ++t) {
            v16h vv = frag_b_f16(sVc, t * 16, 32);
            oacc[t] = wmma_f16(pf, vv, oacc[t]);
        }
        __syncthreads();  // staging of buf^1 done; buf free for next staging
    }

    float* ob = o_ws + ((size_t)b * HW + q0) * DV;
#pragma unroll
    for (int t = 0; t < 8; ++t) {
#pragma unroll
        for (int r = 0; r < 8; ++r) {
            ob[(size_t)(rbase + r) * DV + t * 16 + n] = oacc[t][r] / lrow[r];
        }
    }
}

// ---------------------------------------------------------------------------
// 3) output projection + bias + residual: out = x + o @ Wo + bo
//    wot: [256][128] f16 (Wo^T)
// ---------------------------------------------------------------------------
__global__ __launch_bounds__(256) void outproj_kernel(
    const float* __restrict__ o_ws, const h16* __restrict__ wot,
    const float* __restrict__ bo, const float* __restrict__ x,
    float* __restrict__ out) {
    int wave  = blockIdx.x * (blockDim.x >> 5) + (threadIdx.x >> 5);
    int row0  = wave * 16;
    int lane  = threadIdx.x & 31;
    int rbase = (lane & 16) ? 8 : 0;
    int n     = lane & 15;

    v16h a[4];
#pragma unroll
    for (int kc = 0; kc < 4; ++kc) a[kc] = frag_a_f32(o_ws, row0, DV, kc * 32);

    for (int nt = 0; nt < 16; ++nt) {
        v8f acc = {};
#pragma unroll
        for (int kc = 0; kc < 4; ++kc) {
            v16h bfr = frag_b_f16(wot + kc * 32, nt * 16, DV);
            acc = wmma_f16(a[kc], bfr, acc);
        }
        int ch = nt * 16 + n;
        float bias = bo[ch];
#pragma unroll
        for (int r = 0; r < 8; ++r) {
            size_t idx = (size_t)(row0 + rbase + r) * CIN + ch;
            out[idx] = x[idx] + acc[r] + bias;
        }
    }
}

// ---------------------------------------------------------------------------
// launcher
// ---------------------------------------------------------------------------
extern "C" void kernel_launch(void* const* d_in, const int* in_sizes, int n_in,
                              void* d_out, int out_size, void* d_ws, size_t ws_size,
                              hipStream_t stream) {
    const float* x  = (const float*)d_in[0];
    const float* Wf = (const float*)d_in[1];
    const float* bf = (const float*)d_in[2];
    const float* Wg = (const float*)d_in[3];
    const float* bg = (const float*)d_in[4];
    const float* Wh = (const float*)d_in[5];
    const float* bh = (const float*)d_in[6];
    const float* Wo = (const float*)d_in[7];
    const float* bo = (const float*)d_in[8];
    float* out = (float*)d_out;

    // workspace layout (bytes)
    char* ws = (char*)d_ws;
    size_t off = 0;
    h16* wt_all = (h16*)(ws + off); off += (size_t)192 * 256 * 2;        // 98304
    h16* wot    = (h16*)(ws + off); off += (size_t)256 * 128 * 2;        // +64KB
    h16* fh     = (h16*)(ws + off); off += (size_t)NPIX * DK * 2;        // +4MB
    h16* gh     = (h16*)(ws + off); off += (size_t)NPIX * DK * 2;        // +4MB
    h16* ht     = (h16*)(ws + off); off += (size_t)BATCH * DV * HW * 2;  // +16MB
    float* o_ws = (float*)(ws + off); off += (size_t)NPIX * DV * 4;      // +32MB

    // 0) weight transposes (f32 -> f16, [K][N] -> [N][K])
    transpose_cvt_kernel<<<(256 * 32 + 255) / 256, 256, 0, stream>>>(
        Wf, wt_all, 256, 32);
    transpose_cvt_kernel<<<(256 * 32 + 255) / 256, 256, 0, stream>>>(
        Wg, wt_all + (size_t)32 * 256, 256, 32);
    transpose_cvt_kernel<<<(256 * 128 + 255) / 256, 256, 0, stream>>>(
        Wh, wt_all + (size_t)64 * 256, 256, 128);
    transpose_cvt_kernel<<<(128 * 256 + 255) / 256, 256, 0, stream>>>(
        Wo, wot, 128, 256);

    // 1) projections: 4096 pixel tiles / 8 waves per block
    proj_kernel<<<512, 256, 0, stream>>>(x, wt_all, bf, bg, bh, fh, gh, ht);

    // 2) flash attention: 512 blocks x 8 waves (128 queries per block)
    attn_kernel<<<512, 256, 0, stream>>>(fh, gh, ht, o_ws);

    // 3) output projection + residual
    outproj_kernel<<<512, 256, 0, stream>>>(o_ws, wot, bo, x, out);
}
